// AttentionLSTM_80891414053415
// MI455X (gfx1250) — matrix operand order repaired and checked by reading, not compiled, not run on hardware
//
#include <hip/hip_runtime.h>
#include <hip/hip_bf16.h>

// Problem dims (fixed by reference)
#define Nn 128
#define Tn 256
#define Dn 512
#define Hn 512
#define G4 2048   // 4*H
#define Pp 16     // DA*DA

typedef __attribute__((ext_vector_type(16))) __bf16 v16bf;
typedef __attribute__((ext_vector_type(8)))  float  v8f;
typedef __attribute__((ext_vector_type(8)))  unsigned int v8u;
typedef __attribute__((ext_vector_type(4)))  unsigned int v4u;

__device__ __forceinline__ v8f wmma_bf16(v16bf a, v16bf b, v8f c) {
  // (neg_a, A, neg_b, B, c_mod, C, reuse_a, reuse_b)
  return __builtin_amdgcn_wmma_f32_16x16x32_bf16(false, a, false, b, (short)0, c,
                                                 false, false);
}

__device__ __forceinline__ v16bf load_frag(const __bf16* p) {
  // 16 bf16 = 32B, 32B-aligned in swizzled weight planes
  v8u r = *(const v8u*)p;
  return __builtin_bit_cast(v16bf, r);
}

struct bpair { v4u a; v4u b; };
static_assert(sizeof(bpair) == 32, "bpair must be 32 bytes");

// Build a v16bf A-fragment from two contiguous 8x bf16 (16B) chunks in LDS.
__device__ __forceinline__ v16bf frag2(const __bf16* p0, const __bf16* p1) {
  bpair f;
  f.a = *(const v4u*)p0;
  f.b = *(const v4u*)p1;
  return __builtin_bit_cast(v16bf, f);
}

__device__ __forceinline__ float sigf(float v) {
  return 1.0f / (1.0f + __expf(-v));
}

// ---------------------------------------------------------------------------
// Kernel 0: f32 weights (K=512, Ncol=2048 row-major) -> bf16 hi/lo planes,
// pre-swizzled into WMMA B-fragment order:
//   frag (kb, nt): lane ln holds col nt*16+(ln&15), K = kb*32 + (ln>>4)*8 + e + (e>=8 ? 8 : 0)
// ---------------------------------------------------------------------------
__global__ __launch_bounds__(256) void prep_weights(
    const float* __restrict__ Wx, const float* __restrict__ Wh,
    const float* __restrict__ Wa,
    __bf16* __restrict__ WxHi, __bf16* __restrict__ WxLo,
    __bf16* __restrict__ WhHi, __bf16* __restrict__ WhLo,
    __bf16* __restrict__ WaHi, __bf16* __restrict__ WaLo) {
  int gid = blockIdx.x * 256 + threadIdx.x;    // 3 * 16 * 128 * 32 = 196608
  int w    = gid / (16 * 128 * 32);
  int rem  = gid % (16 * 128 * 32);
  int kb   = rem / (128 * 32);
  int rem2 = rem % (128 * 32);
  int nt   = rem2 / 32;
  int ln   = rem2 % 32;
  const float* W = (w == 0) ? Wx : ((w == 1) ? Wh : Wa);
  __bf16* Hi = (w == 0) ? WxHi : ((w == 1) ? WhHi : WaHi);
  __bf16* Lo = (w == 0) ? WxLo : ((w == 1) ? WhLo : WaLo);
  size_t fo = ((size_t)(kb * 128 + nt) * 32 + ln) * 16;
  int n = nt * 16 + (ln & 15);
  int k0 = kb * 32 + (ln >> 4) * 8;
#pragma unroll
  for (int e = 0; e < 16; ++e) {
    float f = W[(size_t)(k0 + e + (e >= 8 ? 8 : 0)) * G4 + n];
    __bf16 hi = (__bf16)f;
    Hi[fo + e] = hi;
    Lo[fo + e] = (__bf16)(f - (float)hi);
  }
}

// ---------------------------------------------------------------------------
// Kernel 1: Y = x @ Wx + b    (M = N*T = 32768, K = 512, Ncol = 2048)
// One 16x16 tile per wave; split-precision bf16 (3 WMMAs per k-step).
// ---------------------------------------------------------------------------
__global__ __launch_bounds__(256) void gemm_xwx(
    const float* __restrict__ x, const float* __restrict__ bvec,
    const __bf16* __restrict__ WxHi, const __bf16* __restrict__ WxLo,
    float* __restrict__ Y) {
  const int lane = threadIdx.x & 31;
  const int wave = threadIdx.x >> 5;
  const int tile = blockIdx.x * 8 + wave;      // 2048*128 tiles total
  const int tm = tile >> 7;
  const int tn = tile & 127;
  const int m0 = tm * 16, n0 = tn * 16;
  const int lnlo = lane & 15, lnhi = lane >> 4;

  v8f acc;
  {
    float bv = bvec[n0 + lnlo];
#pragma unroll
    for (int r = 0; r < 8; ++r) acc[r] = bv;
  }

  for (int kb = 0; kb < 16; ++kb) {
    const int kbase = kb * 32 + 8 * lnhi;
    const float* xp = x + (size_t)(m0 + lnlo) * Dn + kbase;
    v16bf xhi, xlo;
#pragma unroll
    for (int e = 0; e < 8; ++e) {
      float f1 = xp[e], f2 = xp[16 + e];
      __bf16 h1 = (__bf16)f1, h2 = (__bf16)f2;
      xhi[e] = h1; xhi[e + 8] = h2;
      xlo[e] = (__bf16)(f1 - (float)h1);
      xlo[e + 8] = (__bf16)(f2 - (float)h2);
    }
    const size_t fo = ((size_t)(kb * 128 + tn) * 32 + lane) * 16;
    if (kb < 15) __builtin_prefetch(WxHi + fo + (size_t)128 * 32 * 16, 0, 0);
    v16bf bhi = load_frag(WxHi + fo);
    v16bf blo = load_frag(WxLo + fo);
    acc = wmma_bf16(xhi, bhi, acc);
    acc = wmma_bf16(xhi, blo, acc);
    acc = wmma_bf16(xlo, bhi, acc);
  }
#pragma unroll
  for (int r = 0; r < 8; ++r) {
    Y[(size_t)(m0 + r + 8 * lnhi) * G4 + n0 + lnlo] = acc[r];
  }
}

// ---------------------------------------------------------------------------
// Kernel 2: persistent recurrence. 8 workgroups x 16 batch rows, 512 threads
// (16 waves, 4/SIMD). Each wave owns 2 column-blocks x 4 gates (8 tiles), so
// i/f/o/g for a column stay in one wave. Activations are kept in LDS as
// pre-converted bf16 hi/lo planes in A-fragment chunk order, so the k-loop is
// pure ds_load_b128 + global_load_b128 + v_wmma (no per-fragment cvt chains).
// PRE=true: x@Wx+b read from Y.  PRE=false: fused (3 extra WMMAs/k-step).
// ---------------------------------------------------------------------------
template <bool PRE>
__global__ __launch_bounds__(512) void recurrent(
    const float* __restrict__ x, const float* __restrict__ A,
    const float* __restrict__ bvec,
    const __bf16* __restrict__ WhHi, const __bf16* __restrict__ WhLo,
    const __bf16* __restrict__ WaHi, const __bf16* __restrict__ WaLo,
    const __bf16* __restrict__ WxHi, const __bf16* __restrict__ WxLo,
    const float* __restrict__ Y, float* __restrict__ out) {
  __shared__ float  h32[16][Hn];        // f32 h for scores        (32 KB)
  __shared__ float  c32[16][Hn];        // cell state              (32 KB)
  __shared__ __bf16 hhi[2][16][Hn];     // h hi plane, ping-pong   (32 KB)
  __shared__ __bf16 hlo[2][16][Hn];     // h lo plane, ping-pong   (32 KB)
  __shared__ __bf16 athi[16][Hn];       // attn hi plane           (16 KB)
  __shared__ __bf16 atlo[16][Hn];       // attn lo plane           (16 KB)
  __shared__ float  scp[16][Pp][2];     // score partials
  __shared__ float  sc[16][Pp];         // scores -> softmax weights
  __shared__ float  ew[16][Pp];         // exp weights

  const int tid = threadIdx.x;
  const int lane = tid & 31;
  const int wave = tid >> 5;            // 0..15
  const int rowbase = blockIdx.x * 16;
  const int lnlo = lane & 15, lnhi = lane >> 4;
  const float scale = 0.044194173824159216f;  // 1/sqrt(512)

  // h0 = c0 = mean of A over the 16 attention positions
  for (int idx = tid; idx < 16 * Hn; idx += 512) {
    int n = idx >> 9, hh = idx & 511;
    const float* ap = A + ((size_t)(rowbase + n) * Hn + hh) * Pp;
    float s = 0.f;
#pragma unroll
    for (int p = 0; p < Pp; ++p) s += ap[p];
    s *= (1.0f / 16.0f);
    h32[n][hh] = s;
    c32[n][hh] = s;
    __bf16 hi = (__bf16)s;
    hhi[0][n][hh] = hi;
    hlo[0][n][hh] = (__bf16)(s - (float)hi);
  }
  __syncthreads();

  for (int t = 0; t < Tn; ++t) {
    const int cur = t & 1, nxt = cur ^ 1;

    // ---- 1. scores[n][p] = scale * sum_h h[n][h] * A[n][h][p] ----
    {
      int n = tid >> 5;
      int p = tid & 15;
      int half = (tid >> 4) & 1;
      const float* ap = A + (size_t)(rowbase + n) * Hn * Pp + p;
      float s = 0.f;
      const int h0i = half * 256;
      for (int hh = h0i; hh < h0i + 256; ++hh)
        s = fmaf(h32[n][hh], ap[(size_t)hh * Pp], s);
      scp[n][p][half] = s;
    }
    __syncthreads();
    if (tid < 256) {
      int n = tid >> 4, p = tid & 15;
      sc[n][p] = (scp[n][p][0] + scp[n][p][1]) * scale;
    }
    __syncthreads();

    // ---- 2. softmax over p (P = 16) ----
    {
      int n = (tid >> 4) & 15, p = tid & 15;
      float e = 0.f;
      if (tid < 256) {
        float mx = sc[n][0];
#pragma unroll
        for (int q = 1; q < Pp; ++q) mx = fmaxf(mx, sc[n][q]);
        e = __expf(sc[n][p] - mx);
        ew[n][p] = e;
      }
      __syncthreads();
      if (tid < 256) {
        float sum = 0.f;
#pragma unroll
        for (int q = 0; q < Pp; ++q) sum += ew[n][q];
        sc[n][p] = e / sum;         // sc now holds normalized weights
      }
    }
    __syncthreads();

    // ---- 3. attn[n][h] = sum_p A[n][h][p] * w[n][p] -> bf16 hi/lo ----
    {
      int n = tid >> 5;
      int h0i = (tid & 31) * 16;
      float wl[Pp];
#pragma unroll
      for (int p = 0; p < Pp; ++p) wl[p] = sc[n][p];
      for (int hh = h0i; hh < h0i + 16; ++hh) {
        const float4* ap =
            (const float4*)(A + ((size_t)(rowbase + n) * Hn + hh) * Pp);
        float4 a0 = ap[0], a1 = ap[1], a2 = ap[2], a3 = ap[3];
        float s = a0.x * wl[0] + a0.y * wl[1] + a0.z * wl[2] + a0.w * wl[3] +
                  a1.x * wl[4] + a1.y * wl[5] + a1.z * wl[6] + a1.w * wl[7] +
                  a2.x * wl[8] + a2.y * wl[9] + a2.z * wl[10] + a2.w * wl[11] +
                  a3.x * wl[12] + a3.y * wl[13] + a3.z * wl[14] + a3.w * wl[15];
        __bf16 hi = (__bf16)s;
        athi[n][hh] = hi;
        atlo[n][hh] = (__bf16)(s - (float)hi);
      }
    }
    __syncthreads();

    // ---- 4. per-wave GEMM tiles + gate elementwise ----
    for (int cbi = 0; cbi < 2; ++cbi) {
      const int cb = wave + 16 * cbi;   // column-block 0..31
      const int j0 = cb * 16;
      v8f acc[4];
      if (PRE) {
#pragma unroll
        for (int g = 0; g < 4; ++g) {
          const int col = g * 512 + j0 + lnlo;
#pragma unroll
          for (int r = 0; r < 8; ++r) {
            const int n = rowbase + r + 8 * lnhi;
            acc[g][r] = Y[((size_t)n * Tn + t) * G4 + col];
          }
        }
        if (t + 1 < Tn) {  // pull next step's pre-activations toward L2
          __builtin_prefetch(
              Y + ((size_t)(rowbase + lnlo) * Tn + (t + 1)) * G4 + j0, 0, 0);
        }
      } else {
#pragma unroll
        for (int g = 0; g < 4; ++g) {
          float bv = bvec[g * 512 + j0 + lnlo];
#pragma unroll
          for (int r = 0; r < 8; ++r) acc[g][r] = bv;
        }
      }

      for (int kb = 0; kb < 16; ++kb) {
        const int kbase = kb * 32 + 8 * lnhi;
        v16bf fhhi = frag2(&hhi[cur][lnlo][kbase], &hhi[cur][lnlo][kbase + 16]);
        v16bf fhlo = frag2(&hlo[cur][lnlo][kbase], &hlo[cur][lnlo][kbase + 16]);
        v16bf fahi = frag2(&athi[lnlo][kbase], &athi[lnlo][kbase + 16]);
        v16bf falo = frag2(&atlo[lnlo][kbase], &atlo[lnlo][kbase + 16]);
        v16bf xhi, xlo;
        if (!PRE) {
          const float* xp =
              x + ((size_t)(rowbase + lnlo) * Tn + t) * Dn + kbase;
#pragma unroll
          for (int e = 0; e < 8; ++e) {
            float f1 = xp[e], f2 = xp[16 + e];
            __bf16 b1 = (__bf16)f1, b2 = (__bf16)f2;
            xhi[e] = b1; xhi[e + 8] = b2;
            xlo[e] = (__bf16)(f1 - (float)b1);
            xlo[e + 8] = (__bf16)(f2 - (float)b2);
          }
        }
#pragma unroll
        for (int g = 0; g < 4; ++g) {
          const int nt = (g * 512 + j0) >> 4;  // 0..127
          const size_t fo = ((size_t)(kb * 128 + nt) * 32 + lane) * 16;
          if (kb < 15)
            __builtin_prefetch(WhHi + fo + (size_t)128 * 32 * 16, 0, 0);
          v16bf bhhi = load_frag(WhHi + fo);
          v16bf bhlo = load_frag(WhLo + fo);
          v16bf bahi = load_frag(WaHi + fo);
          v16bf balo = load_frag(WaLo + fo);
          acc[g] = wmma_bf16(fhhi, bhhi, acc[g]);
          acc[g] = wmma_bf16(fhhi, bhlo, acc[g]);
          acc[g] = wmma_bf16(fhlo, bhhi, acc[g]);
          acc[g] = wmma_bf16(fahi, bahi, acc[g]);
          acc[g] = wmma_bf16(fahi, balo, acc[g]);
          acc[g] = wmma_bf16(falo, bahi, acc[g]);
          if (!PRE) {
            v16bf bxhi = load_frag(WxHi + fo);
            v16bf bxlo = load_frag(WxLo + fo);
            acc[g] = wmma_bf16(xhi, bxhi, acc[g]);
            acc[g] = wmma_bf16(xhi, bxlo, acc[g]);
            acc[g] = wmma_bf16(xlo, bxhi, acc[g]);
          }
        }
      }

      // elementwise gates: i=acc[0], f=acc[1], o=acc[2], g=acc[3]
#pragma unroll
      for (int r = 0; r < 8; ++r) {
        const int m = r + 8 * lnhi;
        const int j = j0 + lnlo;
        float iv = acc[0][r], fv = acc[1][r], ov = acc[2][r], gv = acc[3][r];
        float cold = c32[m][j];
        float cn = sigf(fv) * cold + sigf(iv) * tanhf(gv);
        float hn = sigf(ov) * tanhf(cn);
        c32[m][j] = cn;
        h32[m][j] = hn;
        __bf16 hb = (__bf16)hn;
        hhi[nxt][m][j] = hb;
        hlo[nxt][m][j] = (__bf16)(hn - (float)hb);
        out[((size_t)(rowbase + m) * Tn + t) * Hn + j] = hn;
      }
    }
    __syncthreads();
  }
}

// ---------------------------------------------------------------------------
extern "C" void kernel_launch(void* const* d_in, const int* in_sizes, int n_in,
                              void* d_out, int out_size, void* d_ws,
                              size_t ws_size, hipStream_t stream) {
  const float* x  = (const float*)d_in[0];
  const float* A  = (const float*)d_in[1];
  const float* Wx = (const float*)d_in[2];
  const float* Wh = (const float*)d_in[3];
  const float* Wa = (const float*)d_in[4];
  const float* b  = (const float*)d_in[5];
  float* out = (float*)d_out;

  const size_t plane = (size_t)Dn * G4 * sizeof(__bf16);  // 2 MB per plane
  char* ws = (char*)d_ws;
  __bf16* WxHi = (__bf16*)(ws + 0 * plane);
  __bf16* WxLo = (__bf16*)(ws + 1 * plane);
  __bf16* WhHi = (__bf16*)(ws + 2 * plane);
  __bf16* WhLo = (__bf16*)(ws + 3 * plane);
  __bf16* WaHi = (__bf16*)(ws + 4 * plane);
  __bf16* WaLo = (__bf16*)(ws + 5 * plane);
  float* Y = (float*)(ws + 6 * plane);
  const size_t ybytes = (size_t)Nn * Tn * G4 * sizeof(float);  // 256 MB
  const bool pre = ws_size >= 6 * plane + ybytes;

  prep_weights<<<768, 256, 0, stream>>>(Wx, Wh, Wa, WxHi, WxLo, WhHi, WhLo,
                                        WaHi, WaLo);
  if (pre) {
    gemm_xwx<<<(Nn * Tn / 16) * (G4 / 16) / 8, 256, 0, stream>>>(x, b, WxHi,
                                                                 WxLo, Y);
    recurrent<true><<<Nn / 16, 512, 0, stream>>>(x, A, b, WhHi, WhLo, WaHi,
                                                 WaLo, WxHi, WxLo, Y, out);
  } else {
    recurrent<false><<<Nn / 16, 512, 0, stream>>>(x, A, b, WhHi, WhLo, WaHi,
                                                  WaLo, WxHi, WxLo, Y, out);
  }
}
